// SelfAttention_35966056136911
// MI455X (gfx1250) — matrix operand order repaired and checked by reading, not compiled, not run on hardware
//
#include <hip/hip_runtime.h>

// Problem sizes (fixed by reference)
#define BB 8
#define SS 2048
#define DD 512
#define UU 512
#define MM (BB * SS)        // 16384 rows total across batches

typedef __attribute__((ext_vector_type(16))) __bf16 v16bf;
typedef __attribute__((ext_vector_type(8)))  __bf16 v8bf;   // 16 bytes = one b128 load
typedef __attribute__((ext_vector_type(8)))  float  v8f;

union Frag { v16bf v; v8bf h[2]; };

static __device__ __forceinline__ unsigned short f2bf(float f) {
    union { float f; unsigned int u; } x; x.f = f;
    unsigned int u = x.u + 0x7FFFu + ((x.u >> 16) & 1u);   // round-to-nearest-even
    return (unsigned short)(u >> 16);
}

static __device__ __forceinline__ v8f wmma_bf16(const Frag& a, const Frag& b, v8f c) {
    // (neg_a, A, neg_b, B, c_mod, C, reuse_a, reuse_b)
    return __builtin_amdgcn_wmma_f32_16x16x32_bf16(false, a.v, false, b.v,
                                                   (short)0, c, false, false);
}

// ---------------- prep kernels ----------------

__global__ void cvt_f2bf_kernel(const float* __restrict__ src,
                                unsigned short* __restrict__ dst, long long n) {
    long long i = (long long)blockIdx.x * blockDim.x + threadIdx.x;
    long long stride = (long long)gridDim.x * blockDim.x;
    for (; i < n; i += stride) dst[i] = f2bf(src[i]);
}

// per-batch transpose: Xt[b][d][s] = X[b][s][d]  (bf16) -> B-fragments of PV matmul
__global__ void transpose_x_kernel(const float* __restrict__ X,
                                   unsigned short* __restrict__ Xt) {
    long long n = (long long)BB * DD * SS;
    long long i = (long long)blockIdx.x * blockDim.x + threadIdx.x;
    long long stride = (long long)gridDim.x * blockDim.x;
    for (; i < n; i += stride) {
        int b = (int)(i / ((long long)DD * SS));
        int r = (int)(i % ((long long)DD * SS));
        int d = r / SS;
        int s = r % SS;
        Xt[i] = f2bf(X[(long long)b * SS * DD + (long long)s * DD + d]);
    }
}

// Wt[n][k] = W[k][n] (bf16) -> B-fragments of projection matmul
__global__ void transpose_w_kernel(const float* __restrict__ W,
                                   unsigned short* __restrict__ Wt) {
    int i = blockIdx.x * blockDim.x + threadIdx.x;   // exactly DD*UU threads
    int n = i / DD;
    int k = i % DD;
    Wt[i] = f2bf(W[(long long)k * UU + n]);
}

// ---------------- projection: Q = X*Ww + Wb, K = X*Uw + Ub (bf16 out) ----------------
__global__ __launch_bounds__(256) void proj_kernel(
        const unsigned short* __restrict__ Xbf,
        const unsigned short* __restrict__ Wwt,
        const unsigned short* __restrict__ Uwt,
        const float* __restrict__ Wb, const float* __restrict__ Ub,
        unsigned short* __restrict__ Qbf, unsigned short* __restrict__ Kbf) {
    const int gwave = (blockIdx.x * blockDim.x + threadIdx.x) >> 5;
    const int lane  = threadIdx.x & 31;
    const int hi    = lane >> 4;
    const int lo    = lane & 15;
    const int ntile = UU / 16;                       // 32
    const int total = (MM / 16) * ntile;             // 32768 tiles
    if (gwave >= total) return;                      // wave-uniform
    const int m0 = (gwave / ntile) * 16;
    const int n0 = (gwave % ntile) * 16;

    v8f accq = {0.f,0.f,0.f,0.f,0.f,0.f,0.f,0.f};
    v8f acck = accq;

    #pragma unroll 4
    for (int kk = 0; kk < DD; kk += 32) {
        Frag a, bq, bk;
        const unsigned short* ap = Xbf + (size_t)(m0 + lo) * DD + kk + 8 * hi;
        a.h[0] = *(const v8bf*)(ap);
        a.h[1] = *(const v8bf*)(ap + 16);
        const unsigned short* wp = Wwt + (size_t)(n0 + lo) * DD + kk + 8 * hi;
        bq.h[0] = *(const v8bf*)(wp);
        bq.h[1] = *(const v8bf*)(wp + 16);
        const unsigned short* up = Uwt + (size_t)(n0 + lo) * DD + kk + 8 * hi;
        bk.h[0] = *(const v8bf*)(up);
        bk.h[1] = *(const v8bf*)(up + 16);
        accq = wmma_bf16(a, bq, accq);
        acck = wmma_bf16(a, bk, acck);
    }

    const float biasq = Wb[n0 + lo];
    const float biask = Ub[n0 + lo];
    #pragma unroll
    for (int e = 0; e < 8; ++e) {
        size_t ro = (size_t)(m0 + e + 8 * hi) * UU + n0 + lo;   // C/D layout
        Qbf[ro] = f2bf(accq[e] + biasq);
        Kbf[ro] = f2bf(acck[e] + biask);
    }
}

// ---------------- fused flash attention, split-Q ----------------
// One block = 8 waves = 128 query rows; each wave owns 16 rows, full online
// softmax, full 16x512 f32 accumulator (256 VGPRs), and sweeps ALL keys.
// All waves of the WGP stream the SAME K/Xt tiles -> WGP$/L0 hits; no barriers,
// no cross-wave merge. P converts layout through a private LDS region.
__global__ __launch_bounds__(256) void attn_kernel(
        const unsigned short* __restrict__ Qbf,
        const unsigned short* __restrict__ Kbf,
        const unsigned short* __restrict__ Xt,
        float* __restrict__ out) {
    __shared__ unsigned short Pbuf[8][16 * 32];   // per-wave private P tile (bf16)

    const int wave = threadIdx.x >> 5;
    const int lane = threadIdx.x & 31;
    const int hi   = lane >> 4;
    const int lo   = lane & 15;

    const int rows_blk = 128;                                 // 8 waves * 16 rows
    const int b  = (blockIdx.x * rows_blk) / SS;              // batch
    const int q0 = (blockIdx.x * rows_blk) % SS + wave * 16;  // this wave's 16 rows

    const v8f vzero = {0.f,0.f,0.f,0.f,0.f,0.f,0.f,0.f};
    v8f acc[32];                                  // 16 rows x 512 d  (C/D layout per tile)
    #pragma unroll
    for (int i = 0; i < 32; ++i) acc[i] = vzero;

    float mrun[8], lrun[8];
    #pragma unroll
    for (int e = 0; e < 8; ++e) { mrun[e] = -1e30f; lrun[e] = 0.f; }

    const size_t qrow  = (size_t)(b * SS + q0 + lo) * UU;   // Q A-fragment row
    const size_t xbase = (size_t)b * DD * SS;

    for (int c = 0; c < SS / 32; ++c) {           // 64 chunks of 32 keys (full S)
        const int k0 = c * 32;

        // ---- scores tile S[16x32] = Q . K^T over U ----
        v8f s0 = vzero, s1 = vzero;
        #pragma unroll 4
        for (int uu = 0; uu < UU; uu += 32) {
            Frag a, b0, b1;
            const unsigned short* ap = Qbf + qrow + uu + 8 * hi;
            a.h[0] = *(const v8bf*)(ap);
            a.h[1] = *(const v8bf*)(ap + 16);
            const unsigned short* kp0 = Kbf + (size_t)(b * SS + k0 + lo) * UU + uu + 8 * hi;
            b0.h[0] = *(const v8bf*)(kp0);
            b0.h[1] = *(const v8bf*)(kp0 + 16);
            const unsigned short* kp1 = Kbf + (size_t)(b * SS + k0 + 16 + lo) * UU + uu + 8 * hi;
            b1.h[0] = *(const v8bf*)(kp1);
            b1.h[1] = *(const v8bf*)(kp1 + 16);
            s0 = wmma_bf16(a, b0, s0);
            s1 = wmma_bf16(a, b1, s1);
        }

        // ---- online softmax (private to this wave); rows e + 8*hi ----
        float scale[8];
        #pragma unroll
        for (int e = 0; e < 8; ++e) {
            float v0 = s0[e], v1 = s1[e];
            float rm = fmaxf(v0, v1);
            #pragma unroll
            for (int m = 1; m < 16; m <<= 1) rm = fmaxf(rm, __shfl_xor(rm, m, 32));
            float mnew = fmaxf(mrun[e], rm);
            scale[e] = __expf(mrun[e] - mnew);
            float p0 = __expf(v0 - mnew);
            float p1 = __expf(v1 - mnew);
            float ps = p0 + p1;
            #pragma unroll
            for (int m = 1; m < 16; m <<= 1) ps += __shfl_xor(ps, m, 32);
            lrun[e] = lrun[e] * scale[e] + ps;
            mrun[e] = mnew;
            const int row = e + 8 * hi;
            Pbuf[wave][row * 32 + lo]      = f2bf(p0);   // same-wave DS ops are in-order
            Pbuf[wave][row * 32 + 16 + lo] = f2bf(p1);
        }

        // ---- P as A-fragment from private LDS region ----
        Frag pa;
        const unsigned short* pp = &Pbuf[wave][lo * 32 + 8 * hi];
        pa.h[0] = *(const v8bf*)(pp);
        pa.h[1] = *(const v8bf*)(pp + 16);

        // ---- rescale + PV over full d = 512 ----
        #pragma unroll
        for (int nt = 0; nt < 32; ++nt) {
            #pragma unroll
            for (int e = 0; e < 8; ++e) acc[nt][e] *= scale[e];
            Frag bx;
            const unsigned short* xp =
                Xt + xbase + (size_t)(nt * 16 + lo) * SS + k0 + 8 * hi;
            bx.h[0] = *(const v8bf*)(xp);
            bx.h[1] = *(const v8bf*)(xp + 16);
            acc[nt] = wmma_bf16(pa, bx, acc[nt]);
        }
    }

    // ---- normalize with private 1/l and store directly (no merge needed) ----
    float linv[8];
    #pragma unroll
    for (int e = 0; e < 8; ++e) linv[e] = 1.0f / lrun[e];

    #pragma unroll
    for (int nt = 0; nt < 32; ++nt) {
        #pragma unroll
        for (int e = 0; e < 8; ++e) {
            size_t o = (size_t)(b * SS + q0 + e + 8 * hi) * DD + nt * 16 + lo;
            out[o] = acc[nt][e] * linv[e];
        }
    }
}

// ---------------- host launcher ----------------
extern "C" void kernel_launch(void* const* d_in, const int* in_sizes, int n_in,
                              void* d_out, int out_size, void* d_ws, size_t ws_size,
                              hipStream_t stream) {
    const float* X  = (const float*)d_in[0];   // [B,S,D]
    const float* Ww = (const float*)d_in[1];   // [D,U]
    const float* Wb = (const float*)d_in[2];   // [U]
    const float* Uw = (const float*)d_in[3];   // [D,U]
    const float* Ub = (const float*)d_in[4];   // [U]
    float* out = (float*)d_out;                // [B,S,D]

    const size_t nBig = (size_t)MM * DD;       // 8,388,608 elems each
    unsigned short* ws  = (unsigned short*)d_ws;
    unsigned short* Qbf = ws;                  // [M,U]
    unsigned short* Kbf = ws + nBig;           // [M,U]
    unsigned short* Xbf = ws + 2 * nBig;       // [M,D] row-major
    unsigned short* Xt  = ws + 3 * nBig;       // [B,D,S] transposed values
    unsigned short* Wwt = ws + 4 * nBig;       // [U,D]
    unsigned short* Uwt = Wwt + (size_t)UU * DD;

    cvt_f2bf_kernel<<<8192, 256, 0, stream>>>(X, Xbf, (long long)nBig);
    transpose_x_kernel<<<8192, 256, 0, stream>>>(X, Xt);
    transpose_w_kernel<<<(DD * UU) / 256, 256, 0, stream>>>(Ww, Wwt);
    transpose_w_kernel<<<(DD * UU) / 256, 256, 0, stream>>>(Uw, Uwt);

    proj_kernel<<<4096, 256, 0, stream>>>(Xbf, Wwt, Uwt, Wb, Ub, Qbf, Kbf);

    // 128 blocks x 8 waves; each wave owns 16 query rows
    attn_kernel<<<MM / 128, 256, 0, stream>>>(Qbf, Kbf, Xt, out);
}
